// CAModel_30408368455829
// MI455X (gfx1250) — matrix operand order repaired and checked by reading, not compiled, not run on hardware
//
#include <hip/hip_runtime.h>

// NCA update step for MI455X (gfx1250, wave32).
// Kernel 1: perceive (VALU) + GEMM1/GEMM2 via v_wmma_f32_16x16x32_f16 -> x_new in d_ws
// Kernel 2: 3x3 max-pool life masks (old & new alpha) -> out = x_new * mask
// x (64MB) and x_new (64MB) both fit in the 192MB L2, so pass 2 is ~free.

typedef _Float16 half_t;
typedef __attribute__((ext_vector_type(16))) _Float16 v16h;
typedef __attribute__((ext_vector_type(8)))  _Float16 v8h;
typedef __attribute__((ext_vector_type(8)))  float    v8f;

static constexpr int CCH  = 16;    // channels
static constexpr int K3   = 48;    // 3*C perceive features
static constexpr int KPAD = 64;    // pad K to 2 WMMA k-steps
static constexpr int HID  = 128;
static constexpr int BB = 16, HH = 256, WW = 256;
static constexpr int TW   = 64;    // pixels per block (row segment)
static constexpr int HALO = TW + 2;
static constexpr int XSP  = 68;    // padded tile pitch (columns)

__device__ __forceinline__ v16h cat8(v8h lo, v8h hi) {
  return __builtin_shufflevector(lo, hi, 0,1,2,3,4,5,6,7,8,9,10,11,12,13,14,15);
}

__global__ __launch_bounds__(128)
void nca_update_kernel(const float* __restrict__ x,  const float* __restrict__ W1,
                       const float* __restrict__ b1, const float* __restrict__ W2,
                       const float* __restrict__ b2, const float* __restrict__ ru,
                       float* __restrict__ xnew)
{
  __shared__ __align__(16) float  xs[3 * XSP * CCH];      // input tile + halo (f32)
  __shared__ __align__(32) half_t ybuf[TW * KPAD];        // perceive output, A of GEMM1
  __shared__ __align__(32) half_t w1t[HID * KPAD];        // W1^T [d][k], k zero-padded
  __shared__ __align__(32) half_t hbuf[TW * HID];         // hidden, A of GEMM2
  __shared__ __align__(32) half_t w2t[CCH * HID];         // W2^T [c][k]
  __shared__ float b1s[HID];
  __shared__ float b2s[CCH];

  const int tid = threadIdx.x;
  const int nseg = WW / TW;
  int blk  = blockIdx.x;
  int wseg = blk % nseg;  int t2 = blk / nseg;
  int hrow = t2 % HH;     int bidx = t2 / HH;
  int w0   = wseg * TW;

  // ---- stage weights (f32 -> f16, transposed so B-fragments are contiguous K runs)
  for (int i = tid; i < K3 * HID; i += 128) {
    int q = i / HID, d = i % HID;
    w1t[d * KPAD + q] = (half_t)W1[i];
  }
  for (int i = tid; i < (KPAD - K3) * HID; i += 128) {
    int q = K3 + i / HID, d = i % HID;
    w1t[d * KPAD + q] = (half_t)0.0f;          // zero K-pad
  }
  for (int i = tid; i < HID * CCH; i += 128) {
    int k = i / CCH, c = i % CCH;
    w2t[c * HID + k] = (half_t)W2[i];
  }
  for (int i = tid; i < HID; i += 128) b1s[i] = b1[i];
  if (tid < CCH) b2s[tid] = b2[tid];

  // ---- stage x tile with zero-padded halo
  for (int i = tid; i < 3 * HALO * CCH; i += 128) {
    int ch  = i % CCH;  int cr = i / CCH;
    int col = cr % HALO; int r = cr / HALO;
    int gh = hrow + r - 1; int gw = w0 + col - 1;
    float v = 0.0f;
    if (gh >= 0 && gh < HH && gw >= 0 && gw < WW)
      v = x[(((size_t)bidx * HH + gh) * WW + gw) * CCH + ch];
    xs[(r * XSP + col) * CCH + ch] = v;
  }
  // zero K-pad of ybuf (avoid NaN*0 in accumulation)
  for (int i = tid; i < TW * (KPAD - K3); i += 128) {
    int p = i / (KPAD - K3); int k = K3 + i % (KPAD - K3);
    ybuf[p * KPAD + k] = (half_t)0.0f;
  }
  __syncthreads();

  // ---- perceive: identity + sobel-x + sobel-y (13 taps/channel), f32 -> f16
  for (int t = tid; t < TW * CCH; t += 128) {
    int c = t % CCH; int p = t / CCH;
    int col = p + 1;
    const float* r0 = &xs[(0 * XSP + col - 1) * CCH + c];
    const float* r1 = &xs[(1 * XSP + col - 1) * CCH + c];
    const float* r2 = &xs[(2 * XSP + col - 1) * CCH + c];
    float x00 = r0[0], x01 = r0[CCH], x02 = r0[2 * CCH];
    float x10 = r1[0], x11 = r1[CCH], x12 = r1[2 * CCH];
    float x20 = r2[0], x21 = r2[CCH], x22 = r2[2 * CCH];
    float ident = x11;
    float sx = (x02 - x00 + 2.0f * (x12 - x10) + x22 - x20) * 0.125f;
    float sy = (x20 - x00 + 2.0f * (x21 - x01) + x22 - x02) * 0.125f;
    half_t* yp = &ybuf[p * KPAD + 3 * c];
    yp[0] = (half_t)ident; yp[1] = (half_t)sx; yp[2] = (half_t)sy;
  }
  __syncthreads();

  // ---- WMMA GEMMs. wave wv owns pixels [wv*16, wv*16+16)
  const int lane = tid & 31;
  const int wv   = tid >> 5;       // 0..3
  const int m    = lane & 15;      // M row (A) / N col (B,C,D)
  const int g    = lane >> 4;      // lane half
  const int p    = wv * 16 + m;    // pixel owned as A-row

  // A fragments (ISA 7.12.2: lane half g gets K blocks {8g..8g+7, 16+8g..16+8g+7})
  const half_t* yrow = &ybuf[p * KPAD];
  v16h a0 = cat8(*(const v8h*)(yrow +      8 * g), *(const v8h*)(yrow + 16 + 8 * g));
  v16h a1 = cat8(*(const v8h*)(yrow + 32 + 8 * g), *(const v8h*)(yrow + 48 + 8 * g));

  // GEMM1: h = relu(y @ W1 + b1), 8 N-tiles x 2 K-steps
  #pragma unroll
  for (int n = 0; n < 8; n++) {
    const half_t* wrow = &w1t[(n * 16 + m) * KPAD];   // B: N = lane%16, K blocks {8g..8g+7, 16+8g..16+8g+7}
    v16h bA = cat8(*(const v8h*)(wrow +      8 * g), *(const v8h*)(wrow + 16 + 8 * g));
    v16h bB = cat8(*(const v8h*)(wrow + 32 + 8 * g), *(const v8h*)(wrow + 48 + 8 * g));
    v8f acc = {};
    acc = __builtin_amdgcn_wmma_f32_16x16x32_f16(false, a0, false, bA, (short)0, acc, false, false);
    acc = __builtin_amdgcn_wmma_f32_16x16x32_f16(false, a1, false, bB, (short)0, acc, false, false);
    float bias = b1s[n * 16 + m];
    #pragma unroll
    for (int r = 0; r < 8; r++) {                      // D: M = r + 8g, N = m
      float v = acc[r] + bias;
      v = v > 0.0f ? v : 0.0f;
      hbuf[(wv * 16 + r + 8 * g) * HID + n * 16 + m] = (half_t)v;
    }
  }
  __syncthreads();

  // GEMM2: dx = h @ W2 + b2, 4 K-steps, N = 16 channels
  v8f dacc = {};
  #pragma unroll
  for (int f = 0; f < 4; f++) {
    const half_t* hr = &hbuf[p * HID + f * 32];
    v16h ah = cat8(*(const v8h*)(hr + 8 * g), *(const v8h*)(hr + 16 + 8 * g));
    const half_t* w2r = &w2t[m * HID + f * 32];
    v16h bh = cat8(*(const v8h*)(w2r + 8 * g), *(const v8h*)(w2r + 16 + 8 * g));
    dacc = __builtin_amdgcn_wmma_f32_16x16x32_f16(false, ah, false, bh, (short)0, dacc, false, false);
  }

  // ---- epilogue: x_new = x + dx * (rand <= 0.5)   (STEP_SIZE = 1)
  float bias2 = b2s[m];
  size_t rowbase = ((size_t)bidx * HH + hrow) * WW + w0;
  #pragma unroll
  for (int r = 0; r < 8; r++) {
    int pp = wv * 16 + r + 8 * g;                      // D: M = r + 8g
    float dx = dacc[r] + bias2;
    float um = (ru[rowbase + pp] <= 0.5f) ? 1.0f : 0.0f;
    float xo = xs[(XSP + (pp + 1)) * CCH + m];
    xnew[(rowbase + pp) * CCH + m] = xo + dx * um;
  }
}

__device__ __forceinline__ float max9(const float* a, int col) {
  float m0 = fmaxf(fmaxf(a[col - 1],           a[col]),           a[col + 1]);
  float m1 = fmaxf(fmaxf(a[XSP + col - 1],     a[XSP + col]),     a[XSP + col + 1]);
  float m2 = fmaxf(fmaxf(a[2 * XSP + col - 1], a[2 * XSP + col]), a[2 * XSP + col + 1]);
  return fmaxf(fmaxf(m0, m1), m2);
}

__global__ __launch_bounds__(256)
void nca_mask_kernel(const float* __restrict__ x, const float* __restrict__ xnew,
                     float* __restrict__ out)
{
  __shared__ float aold[3 * XSP];
  __shared__ float anew[3 * XSP];
  __shared__ float life[TW];

  const int tid = threadIdx.x;
  const int nseg = WW / TW;
  int blk  = blockIdx.x;
  int wseg = blk % nseg;  int t2 = blk / nseg;
  int hrow = t2 % HH;     int bidx = t2 / HH;
  int w0   = wseg * TW;

  for (int i = tid; i < 3 * HALO; i += 256) {
    int col = i % HALO; int r = i / HALO;
    int gh = hrow + r - 1; int gw = w0 + col - 1;
    float ao = -3.0e38f, an = -3.0e38f;                // -inf padding for max-pool
    if (gh >= 0 && gh < HH && gw >= 0 && gw < WW) {
      size_t base = (((size_t)bidx * HH + gh) * WW + gw) * CCH + 3;  // alpha channel
      ao = x[base]; an = xnew[base];
    }
    aold[r * XSP + col] = ao; anew[r * XSP + col] = an;
  }
  __syncthreads();

  if (tid < TW) {
    int col = tid + 1;
    life[tid] = (max9(aold, col) > 0.1f && max9(anew, col) > 0.1f) ? 1.0f : 0.0f;
  }
  __syncthreads();

  size_t rowbase = ((size_t)bidx * HH + hrow) * WW + w0;
  for (int t = tid; t < TW * CCH; t += 256) {
    int c = t % CCH; int p = t / CCH;
    size_t idx = (rowbase + p) * CCH + c;
    out[idx] = xnew[idx] * life[p];
  }
}

extern "C" void kernel_launch(void* const* d_in, const int* in_sizes, int n_in,
                              void* d_out, int out_size, void* d_ws, size_t ws_size,
                              hipStream_t stream) {
  const float* x  = (const float*)d_in[0];
  const float* W1 = (const float*)d_in[1];
  const float* b1 = (const float*)d_in[2];
  const float* W2 = (const float*)d_in[3];
  const float* b2 = (const float*)d_in[4];
  const float* ru = (const float*)d_in[5];
  float* out  = (float*)d_out;
  float* xnew = (float*)d_ws;   // B*H*W*C f32 = 64 MB scratch

  int nblocks = BB * HH * (WW / TW);   // 16384
  nca_update_kernel<<<nblocks, 128, 0, stream>>>(x, W1, b1, W2, b2, ru, xnew);
  nca_mask_kernel<<<nblocks, 256, 0, stream>>>(x, xnew, out);
}
